// Autoformer_36773509988387
// MI455X (gfx1250) — matrix operand order repaired and checked by reading, not compiled, not run on hardware
//
#include <hip/hip_runtime.h>
#include <cstddef>

// ---------------------------------------------------------------------------
// Autoformer forward for gfx1250 (MI455X).
// GEMMs: bf16 operands, f32 accumulate via V_WMMA_F32_16X16X32_BF16.
// ---------------------------------------------------------------------------

typedef __bf16 v16bf __attribute__((ext_vector_type(16)));
typedef __bf16 v8bf  __attribute__((ext_vector_type(8)));
typedef float  v8f   __attribute__((ext_vector_type(8)));

#define BATCH 32
#define SEQL  1024
#define DMODEL 512
#define DFF   2048
#define EIN   21
#define NTOPK 20
#define MAPAD 12         // (25-1)/2
#define PREDL 512

static __device__ __forceinline__ float gelu_f(float x) {
  return 0.5f * x * (1.0f + erff(x * 0.7071067811865475f));
}

// ---------------- GEMM: C[M,N] = A[M,K](bf16) @ Bt[N,K](bf16)^T -------------
// block = 128 threads (4 waves); each wave computes a 16x32 tile (2 accums).
// Optional epilogue: +bias, gelu, +residual; writes f32 C and/or bf16 Cbf.
__global__ void k_gemm_bt(const __bf16* __restrict__ A,
                          const __bf16* __restrict__ Bt,
                          const float* __restrict__ bias,
                          const float* __restrict__ res,
                          float* __restrict__ C,
                          __bf16* __restrict__ Cbf,
                          int M, int N, int K, int gelu,
                          size_t sA, size_t sB, size_t sC) {
  const int bz   = blockIdx.z;
  const __bf16* Ab = A  + sA * bz;
  const __bf16* Bb = Bt + sB * bz;
  const int wave = threadIdx.x >> 5;
  const int lane = threadIdx.x & 31;
  const int lm   = lane & 15;
  const int hi   = lane >> 4;
  const int m0   = blockIdx.y << 4;
  const int n0   = (blockIdx.x << 7) + (wave << 5);
  const __bf16* Arow  = Ab + (size_t)(m0 + lm) * K;
  const __bf16* Brow0 = Bb + (size_t)(n0 + lm) * K;
  const __bf16* Brow1 = Bb + (size_t)(n0 + 16 + lm) * K;
  const int aoff0 = hi * 16;       // contiguous k = 16*hi + s, matches B
  const int aoff1 = hi * 16 + 8;
  const int boff  = hi * 16;       // contiguous k = 16*hi + s, same as A

  v8f acc0 = {};
  v8f acc1 = {};
  for (int k = 0; k < K; k += 32) {
    v8bf a0 = *reinterpret_cast<const v8bf*>(Arow + k + aoff0);
    v8bf a1 = *reinterpret_cast<const v8bf*>(Arow + k + aoff1);
    v16bf a = __builtin_shufflevector(a0, a1, 0, 1, 2, 3, 4, 5, 6, 7,
                                      8, 9, 10, 11, 12, 13, 14, 15);
    v16bf b0 = *reinterpret_cast<const v16bf*>(Brow0 + k + boff);
    v16bf b1 = *reinterpret_cast<const v16bf*>(Brow1 + k + boff);
    acc0 = __builtin_amdgcn_wmma_f32_16x16x32_bf16(false, a, false, b0,
                                                   (short)0, acc0, false, false);
    acc1 = __builtin_amdgcn_wmma_f32_16x16x32_bf16(false, a, false, b1,
                                                   (short)0, acc1, false, false);
  }
  const int rowBase = m0 + (hi << 3);
  const int col     = n0 + lm;
#pragma unroll
  for (int r = 0; r < 8; ++r) {
    size_t i0 = sC * bz + (size_t)(rowBase + r) * N + col;
    float v0 = acc0[r], v1 = acc1[r];
    if (bias) { v0 += bias[col]; v1 += bias[col + 16]; }
    if (gelu) { v0 = gelu_f(v0); v1 = gelu_f(v1); }
    if (res)  { v0 += res[i0];   v1 += res[i0 + 16]; }
    if (C)   { C[i0] = v0;              C[i0 + 16] = v1; }
    if (Cbf) { Cbf[i0] = (__bf16)v0;    Cbf[i0 + 16] = (__bf16)v1; }
  }
}

// ---------------- f32 -> bf16 elementwise -----------------------------------
__global__ void k_cvt(const float* __restrict__ src, __bf16* __restrict__ dst,
                      size_t n) {
  size_t i = (size_t)blockIdx.x * 256 + threadIdx.x;
  if (i < n) dst[i] = (__bf16)src[i];
}

// ---------------- weight f32 [K,N] -> bf16 transposed [N,K] -----------------
__global__ void k_wcvt(const float* __restrict__ W, __bf16* __restrict__ Wt,
                       int K, int N) {
  size_t idx = (size_t)blockIdx.x * 256 + threadIdx.x;
  if (idx >= (size_t)K * N) return;
  int k = (int)(idx / N), n = (int)(idx % N);
  Wt[(size_t)n * K + k] = (__bf16)W[idx];
}

// ---------------- corr[b,tau] = (1/C) sum_t S[b,t,(t-tau)%L] ----------------
__global__ void k_corr(const float* __restrict__ S, float* __restrict__ corr,
                       int L, float inv) {
  const int b = blockIdx.x;
  const int tau = threadIdx.x;            // blockDim.x == L == 1024
  const float* Sb = S + (size_t)b * L * L;
  float acc = 0.0f;
  for (int t = 0; t < L; ++t)
    acc += Sb[(size_t)t * L + ((t - tau) & (L - 1))];
  corr[(size_t)b * L + tau] = acc * inv;
}

// ---------------- top-20 + softmax ------------------------------------------
__global__ void k_topk(const float* __restrict__ corr, float* __restrict__ wout,
                       int* __restrict__ dout, int L) {
  const int b = blockIdx.x;
  const int tid = threadIdx.x;            // 256
  __shared__ float vals[SEQL];
  __shared__ float rv[256];
  __shared__ int   ri[256];
  __shared__ float selv[NTOPK];
  __shared__ int   seli[NTOPK];
  for (int j = tid; j < L; j += 256) vals[j] = corr[(size_t)b * L + j];
  __syncthreads();
  for (int it = 0; it < NTOPK; ++it) {
    float bv = -3.4e38f; int bi = 0;
    for (int j = tid; j < L; j += 256) {
      float v = vals[j];
      if (v > bv) { bv = v; bi = j; }
    }
    rv[tid] = bv; ri[tid] = bi;
    __syncthreads();
    for (int s = 128; s > 0; s >>= 1) {
      if (tid < s) {
        if (rv[tid + s] > rv[tid] ||
            (rv[tid + s] == rv[tid] && ri[tid + s] < ri[tid])) {
          rv[tid] = rv[tid + s]; ri[tid] = ri[tid + s];
        }
      }
      __syncthreads();
    }
    if (tid == 0) { selv[it] = rv[0]; seli[it] = ri[0]; vals[ri[0]] = -3.4e38f; }
    __syncthreads();
  }
  if (tid == 0) {
    float mx = selv[0], s = 0.0f, e[NTOPK];
    for (int i = 0; i < NTOPK; ++i) { e[i] = __expf(selv[i] - mx); s += e[i]; }
    for (int i = 0; i < NTOPK; ++i) {
      wout[b * NTOPK + i] = e[i] / s;
      dout[b * NTOPK + i] = seli[i];
    }
  }
}

// ---------------- agg[b,t,c] = sum_i w_i * V[b,(t+delay_i)%L,c] -------------
__global__ void k_agg(const float* __restrict__ V, const float* __restrict__ w,
                      const int* __restrict__ delay, float* __restrict__ out,
                      int L, int D) {
  const int bt = blockIdx.x;
  const int b = bt / L, t = bt % L;
  __shared__ float sw[NTOPK];
  __shared__ int sd[NTOPK];
  if (threadIdx.x < NTOPK) {
    sw[threadIdx.x] = w[b * NTOPK + threadIdx.x];
    sd[threadIdx.x] = delay[b * NTOPK + threadIdx.x];
  }
  __syncthreads();
  const int c = threadIdx.x;              // 512
  float acc = 0.0f;
#pragma unroll
  for (int i = 0; i < NTOPK; ++i)
    acc += sw[i] * V[((size_t)b * L + ((t + sd[i]) & (L - 1))) * D + c];
  out[(size_t)bt * D + c] = acc;
}

// ---------------- series decomposition on D=512 channels --------------------
__global__ void k_decomp(const float* __restrict__ in, float* __restrict__ seas,
                         float* __restrict__ trend_acc, int L, int D) {
  const int bt = blockIdx.x;
  const int b = bt / L, t = bt % L;
  const int c = threadIdx.x;              // 512
  const float* base = in + (size_t)b * L * D;
  float s = 0.0f;
#pragma unroll
  for (int j = -MAPAD; j <= MAPAD; ++j) {
    int tt = t + j;
    tt = tt < 0 ? 0 : (tt >= L ? L - 1 : tt);
    s += base[(size_t)tt * D + c];
  }
  const float mm = s * (1.0f / 25.0f);
  const size_t idx = (size_t)bt * D + c;
  seas[idx] = in[idx] - mm;
  if (trend_acc) trend_acc[idx] += mm;
}

// ---------------- circular token conv (21->512) + positional encoding -------
__global__ void k_embed(const float* __restrict__ x, const float* __restrict__ w,
                        float* __restrict__ out, int L) {
  const int bt = blockIdx.x;
  const int b = bt / L, t = bt % L;
  __shared__ float xs[3 * EIN];
  const int tid = threadIdx.x;            // 512
  if (tid < 3 * EIN) {
    int k = tid / EIN, i = tid % EIN;
    int row = (t - 1 + k + L) % L;
    xs[k * EIN + i] = x[((size_t)b * L + row) * EIN + i];
  }
  __syncthreads();
  const int d = tid;
  const float* wd = w + (size_t)d * EIN * 3;
  float acc = 0.0f;
#pragma unroll
  for (int i = 0; i < EIN; ++i) {
    acc += xs[0 * EIN + i] * wd[i * 3 + 0];
    acc += xs[1 * EIN + i] * wd[i * 3 + 1];
    acc += xs[2 * EIN + i] * wd[i * 3 + 2];
  }
  const int half2 = (d >> 1) << 1;
  const float freq = __expf(-(9.210340371976184f / 512.0f) * (float)half2);
  const float ang = (float)t * freq;
  acc += (d & 1) ? __cosf(ang) : __sinf(ang);
  out[(size_t)bt * DMODEL + d] = acc;
}

// ---------------- row layernorm (over D) ------------------------------------
__global__ void k_ln(const float* __restrict__ in, const float* __restrict__ g,
                     const float* __restrict__ bb, float* __restrict__ out,
                     int D) {
  __shared__ float s1[512];
  __shared__ float s2[512];
  const size_t row = blockIdx.x;
  const int c = threadIdx.x;              // 512
  const float x = in[row * D + c];
  s1[c] = x; s2[c] = x * x;
  __syncthreads();
  for (int s = 256; s > 0; s >>= 1) {
    if (c < s) { s1[c] += s1[c + s]; s2[c] += s2[c + s]; }
    __syncthreads();
  }
  const float mu = s1[0] / (float)D;
  const float var = s2[0] / (float)D - mu * mu;
  out[row * D + c] = (x - mu) * rsqrtf(var + 1e-5f) * g[c] + bb[c];
}

// ---------------- subtract per-(b,c) mean over time axis (in place) ---------
__global__ void k_subcol(float* __restrict__ x, int L, int D) {
  const int b = blockIdx.x;
  const int c = threadIdx.x;              // 512
  float* base = x + (size_t)b * L * D;
  float s = 0.0f;
  for (int t = 0; t < L; ++t) s += base[(size_t)t * D + c];
  const float m = s / (float)L;
  for (int t = 0; t < L; ++t) base[(size_t)t * D + c] -= m;
}

// ---------------- trend circular conv (512->21), accumulate into trend ------
__global__ void k_tconv(const float* __restrict__ ts, const float* __restrict__ w,
                        float* __restrict__ trend, int L, int D) {
  const int bt = blockIdx.x;
  const int b = bt / L, t = bt % L;
  const int tid = threadIdx.x;            // 256
  const float* base = ts + (size_t)b * L * D;
  const int r0 = (t - 1 + L) % L, r1 = t, r2 = (t + 1) % L;
  float part[EIN];
#pragma unroll
  for (int o = 0; o < EIN; ++o) part[o] = 0.0f;
  for (int c = tid; c < D; c += 256) {
    float x0 = base[(size_t)r0 * D + c];
    float x1 = base[(size_t)r1 * D + c];
    float x2 = base[(size_t)r2 * D + c];
#pragma unroll
    for (int o = 0; o < EIN; ++o) {
      const float* wp = w + ((size_t)o * D + c) * 3;
      part[o] += x0 * wp[0] + x1 * wp[1] + x2 * wp[2];
    }
  }
  __shared__ float red[256];
  for (int o = 0; o < EIN; ++o) {
    red[tid] = part[o];
    __syncthreads();
    for (int s = 128; s > 0; s >>= 1) {
      if (tid < s) red[tid] += red[tid + s];
      __syncthreads();
    }
    if (tid == 0) trend[(size_t)bt * EIN + o] += red[0];
    __syncthreads();
  }
}

// ---------------- final projection + trend projection + slice ---------------
__global__ void k_final(const float* __restrict__ dec, const float* __restrict__ trend,
                        const float* __restrict__ pw, const float* __restrict__ pb,
                        const float* __restrict__ tp, float* __restrict__ out,
                        int L, int D, int PRED) {
  const int bt = blockIdx.x;              // b*PRED + tt
  const int b = bt / PRED, tt = bt % PRED;
  const int t = tt + (L - PRED);
  __shared__ float row[DMODEL];
  __shared__ float trow[EIN];
  const int tid = threadIdx.x;            // 128
  for (int j = tid; j < D; j += 128) row[j] = dec[((size_t)b * L + t) * D + j];
  if (tid < EIN) trow[tid] = trend[((size_t)b * L + t) * EIN + tid];
  __syncthreads();
  if (tid < EIN) {
    float acc = pb[tid];
    for (int c = 0; c < D; ++c) acc += row[c] * pw[c * EIN + tid];
#pragma unroll
    for (int i = 0; i < EIN; ++i) acc += trow[i] * tp[i * EIN + tid];
    out[(size_t)bt * EIN + tid] = acc;
  }
}

// ---------------- small helpers: mean over L, 21-ch decomp, init build ------
__global__ void k_meanenc(const float* __restrict__ x, float* __restrict__ mv,
                          int L) {
  const int b = blockIdx.x;
  const int c = threadIdx.x;
  if (c >= EIN) return;
  float s = 0.0f;
  for (int t = 0; t < L; ++t) s += x[((size_t)b * L + t) * EIN + c];
  mv[b * EIN + c] = s / (float)L;
}

__global__ void k_decomp_s(const float* __restrict__ x, float* __restrict__ seas,
                           float* __restrict__ tr, int L) {
  const int bt = blockIdx.x;
  const int b = bt / L, t = bt % L;
  const int c = threadIdx.x;
  if (c >= EIN) return;
  const float* base = x + (size_t)b * L * EIN;
  float s = 0.0f;
#pragma unroll
  for (int j = -MAPAD; j <= MAPAD; ++j) {
    int tt = t + j;
    tt = tt < 0 ? 0 : (tt >= L ? L - 1 : tt);
    s += base[(size_t)tt * EIN + c];
  }
  const float mm = s * (1.0f / 25.0f);
  seas[(size_t)bt * EIN + c] = x[(size_t)bt * EIN + c] - mm;
  tr[(size_t)bt * EIN + c] = mm;
}

__global__ void k_build(const float* __restrict__ seasf, const float* __restrict__ trf,
                        const float* __restrict__ mv, float* __restrict__ sinit,
                        float* __restrict__ tinit, int L, int LABEL) {
  const int bt = blockIdx.x;
  const int b = bt / L, t = bt % L;
  const int c = threadIdx.x;
  if (c >= EIN) return;
  const size_t idx = (size_t)bt * EIN + c;
  if (t < LABEL) {
    const int src = t + (L - LABEL);
    sinit[idx] = seasf[((size_t)b * L + src) * EIN + c];
    tinit[idx] = trf[((size_t)b * L + src) * EIN + c];
  } else {
    sinit[idx] = 0.0f;
    tinit[idx] = mv[b * EIN + c];
  }
}

// ===========================================================================
extern "C" void kernel_launch(void* const* d_in, const int* in_sizes, int n_in,
                              void* d_out, int out_size, void* d_ws, size_t ws_size,
                              hipStream_t stream) {
  (void)in_sizes; (void)n_in; (void)out_size; (void)ws_size;
  const float* in[49];
  for (int i = 0; i < 49; ++i) in[i] = (const float*)d_in[i];

  const float* x_enc   = in[0];
  const float* enc_tok = in[1];
  const float* dec_tok = in[2];
  const float* enc_g = in[23]; const float* enc_b = in[24];
  const float* dec_w1f = in[41]; const float* dec_w2f = in[42];
  const float* wtrend = in[43];
  const float* dec_g = in[44]; const float* dec_b = in[45];
  const float* proj_w = in[46]; const float* proj_b = in[47];
  const float* tproj_w = in[48];

  // ----- workspace layout -----
  constexpr size_t ACT = (size_t)BATCH * SEQL * DMODEL;   // 16,777,216
  float* ws = (float*)d_ws;
  size_t off = 0;
  auto allocF = [&](size_t n) {
    float* p = ws + off; off += (n + 63) & ~(size_t)63; return p;
  };
  float* X     = allocF(ACT);
  float* T     = allocF(ACT);
  float* V     = allocF(ACT);
  float* AGG   = allocF(ACT);
  float* ENC   = allocF(ACT);
  float* TSUM  = allocF(ACT);
  float* S     = allocF((size_t)8 * SEQL * SEQL);
  float* CORR  = allocF((size_t)BATCH * SEQL);
  float* Wts   = allocF(BATCH * NTOPK);
  int*   Dly   = (int*)allocF(BATCH * NTOPK);
  float* SEASF = allocF((size_t)BATCH * SEQL * EIN);
  float* TRF   = allocF((size_t)BATCH * SEQL * EIN);
  float* SINIT = allocF((size_t)BATCH * SEQL * EIN);
  float* TRENDB= allocF((size_t)BATCH * SEQL * EIN);
  float* MV    = allocF(BATCH * EIN);

  __bf16* bp = (__bf16*)(ws + off);
  size_t hoff = 0;
  auto allocH = [&](size_t n) {
    __bf16* p = bp + hoff; hoff += (n + 63) & ~(size_t)63; return p;
  };
  __bf16* Xbf   = allocH(ACT);
  __bf16* Qbf   = allocH(ACT);
  __bf16* Kbf   = allocH(ACT);
  __bf16* FFbf  = allocH(ACT);          // also AGGbf (disjoint phases)
  __bf16* AGGbf = FFbf;
  __bf16* ENCbf = allocH(ACT);

  const int M = BATCH * SEQL;           // 32768

  // ----- transpose-convert all GEMM weights to bf16 [N,K] -----
  auto wcvt = [&](const float* W, int K, int N) {
    __bf16* p = allocH((size_t)K * N);
    size_t tot = (size_t)K * N;
    k_wcvt<<<(unsigned)((tot + 255) / 256), 256, 0, stream>>>(W, p, K, N);
    return p;
  };
  struct APb { const __bf16 *wq, *wk, *wv, *wo;
               const float *bq, *bk, *bv, *bo; };
  auto mk_attn = [&](int base) {
    APb p;
    p.wq = wcvt(in[base + 0], DMODEL, DMODEL); p.bq = in[base + 1];
    p.wk = wcvt(in[base + 2], DMODEL, DMODEL); p.bk = in[base + 3];
    p.wv = wcvt(in[base + 4], DMODEL, DMODEL); p.bv = in[base + 5];
    p.wo = wcvt(in[base + 6], DMODEL, DMODEL); p.bo = in[base + 7];
    return p;
  };
  APb enc_attn[2];
  const __bf16* enc_w1[2]; const __bf16* enc_w2[2];
  for (int l = 0; l < 2; ++l) {
    enc_attn[l] = mk_attn(3 + l * 10);
    enc_w1[l] = wcvt(in[3 + l * 10 + 8], DMODEL, DFF);
    enc_w2[l] = wcvt(in[3 + l * 10 + 9], DFF, DMODEL);
  }
  APb dec_self  = mk_attn(25);
  APb dec_cross = mk_attn(33);
  const __bf16* dec_w1 = wcvt(dec_w1f, DMODEL, DFF);
  const __bf16* dec_w2 = wcvt(dec_w2f, DFF, DMODEL);

  // ----- launch helpers -----
  auto gemm = [&](const __bf16* A, const __bf16* Bt, const float* bias,
                  const float* res, float* C, __bf16* Cbf,
                  int Mm, int N, int K, int gelu) {
    dim3 g(N / 128, Mm / 16, 1);
    k_gemm_bt<<<g, 128, 0, stream>>>(A, Bt, bias, res, C, Cbf,
                                     Mm, N, K, gelu, 0, 0, 0);
  };
  auto cvt = [&](const float* s, __bf16* d, size_t n) {
    k_cvt<<<(unsigned)((n + 255) / 256), 256, 0, stream>>>(s, d, n);
  };

  auto attn = [&](const __bf16* xq, const __bf16* xkv, const APb& p,
                  float* out, const float* resid) {
    gemm(xq,  p.wq, p.bq, nullptr, nullptr, Qbf, M, DMODEL, DMODEL, 0);
    gemm(xkv, p.wk, p.bk, nullptr, nullptr, Kbf, M, DMODEL, DMODEL, 0);
    gemm(xkv, p.wv, p.bv, nullptr, V, nullptr, M, DMODEL, DMODEL, 0);
    for (int cb = 0; cb < BATCH; cb += 8) {
      dim3 g(SEQL / 128, SEQL / 16, 8);
      k_gemm_bt<<<g, 128, 0, stream>>>(Qbf + (size_t)cb * SEQL * DMODEL,
                                       Kbf + (size_t)cb * SEQL * DMODEL,
                                       nullptr, nullptr, S, nullptr,
                                       SEQL, SEQL, DMODEL, 0,
                                       (size_t)SEQL * DMODEL,
                                       (size_t)SEQL * DMODEL,
                                       (size_t)SEQL * SEQL);
      k_corr<<<8, SEQL, 0, stream>>>(S, CORR + (size_t)cb * SEQL,
                                     SEQL, 1.0f / (float)DMODEL);
    }
    k_topk<<<BATCH, 256, 0, stream>>>(CORR, Wts, Dly, SEQL);
    k_agg<<<M, DMODEL, 0, stream>>>(V, Wts, Dly, AGG, SEQL, DMODEL);
    cvt(AGG, AGGbf, ACT);
    gemm(AGGbf, p.wo, p.bo, resid, out, nullptr, M, DMODEL, DMODEL, 0);
  };

  auto ffn = [&](const __bf16* w1, const __bf16* w2) {
    for (int mc = 0; mc < 4; ++mc) {
      const size_t off1 = (size_t)mc * 8192 * DMODEL;
      gemm(Xbf + off1, w1, nullptr, nullptr, nullptr, FFbf, 8192, DFF, DMODEL, 1);
      gemm(FFbf, w2, nullptr, X + off1, T + off1, nullptr, 8192, DMODEL, DFF, 0);
    }
  };

  // ----- preprocessing -----
  k_meanenc<<<BATCH, 32, 0, stream>>>(x_enc, MV, SEQL);
  k_decomp_s<<<M, 32, 0, stream>>>(x_enc, SEASF, TRF, SEQL);
  k_build<<<M, 32, 0, stream>>>(SEASF, TRF, MV, SINIT, TRENDB, SEQL, PREDL);

  // ----- encoder -----
  k_embed<<<M, DMODEL, 0, stream>>>(x_enc, enc_tok, X, SEQL);
  cvt(X, Xbf, ACT);
  for (int l = 0; l < 2; ++l) {
    attn(Xbf, Xbf, enc_attn[l], T, X);
    k_decomp<<<M, DMODEL, 0, stream>>>(T, X, nullptr, SEQL, DMODEL);
    cvt(X, Xbf, ACT);
    ffn(enc_w1[l], enc_w2[l]);
    k_decomp<<<M, DMODEL, 0, stream>>>(T, X, nullptr, SEQL, DMODEL);
    cvt(X, Xbf, ACT);
  }
  k_ln<<<M, DMODEL, 0, stream>>>(X, enc_g, enc_b, ENC, DMODEL);
  k_subcol<<<BATCH, DMODEL, 0, stream>>>(ENC, SEQL, DMODEL);
  cvt(ENC, ENCbf, ACT);

  // ----- decoder -----
  k_embed<<<M, DMODEL, 0, stream>>>(SINIT, dec_tok, X, SEQL);
  cvt(X, Xbf, ACT);
  (void)hipMemsetAsync(TSUM, 0, ACT * sizeof(float), stream);

  attn(Xbf, Xbf, dec_self, T, X);
  k_decomp<<<M, DMODEL, 0, stream>>>(T, X, TSUM, SEQL, DMODEL);
  cvt(X, Xbf, ACT);
  attn(Xbf, ENCbf, dec_cross, T, X);
  k_decomp<<<M, DMODEL, 0, stream>>>(T, X, TSUM, SEQL, DMODEL);
  cvt(X, Xbf, ACT);
  ffn(dec_w1, dec_w2);
  k_decomp<<<M, DMODEL, 0, stream>>>(T, X, TSUM, SEQL, DMODEL);
  k_tconv<<<M, 256, 0, stream>>>(TSUM, wtrend, TRENDB, SEQL, DMODEL);

  k_ln<<<M, DMODEL, 0, stream>>>(X, dec_g, dec_b, T, DMODEL);
  k_subcol<<<BATCH, DMODEL, 0, stream>>>(T, SEQL, DMODEL);

  k_final<<<BATCH * PREDL, 128, 0, stream>>>(T, TRENDB, proj_w, proj_b, tproj_w,
                                             (float*)d_out, SEQL, DMODEL, PREDL);
}
